// FullAttention_22488448762488
// MI455X (gfx1250) — matrix operand order repaired and checked by reading, not compiled, not run on hardware
//
#include <hip/hip_runtime.h>
#include <hip/hip_bf16.h>

typedef _Float16 h16v __attribute__((ext_vector_type(16)));
typedef _Float16 h8v  __attribute__((ext_vector_type(8)));
typedef float    f8v  __attribute__((ext_vector_type(8)));

#define B_    2
#define LIMG_ 2048
#define LTXT_ 256
#define DIM_  1024
#define NH_   16
#define HD_   64
#define S_    (LIMG_ + LTXT_)   // 2304
#define MROWS (B_ * S_)         // 4608
#define QTILES (S_ / 16)        // 144
#define SCALE_ 0.125f           // HD^-0.5

__device__ __forceinline__ f8v wmma_f16(h16v a, h16v b, f8v c) {
    return __builtin_amdgcn_wmma_f32_16x16x32_f16(
        /*neg_a=*/false, a, /*neg_b=*/false, b,
        /*c_mod=*/(short)0, c, /*reuse_a=*/false, /*reuse_b=*/false);
}

__device__ __forceinline__ h16v make_afrag(const _Float16* rowptr, int koff, int hi8) {
    // A 16x32 f16 layout: lane holds row M=lane%16; k = koff + 16*hi + {0..15}
    h8v lo = *(const h8v*)(rowptr + koff + 2 * hi8);
    h8v hv = *(const h8v*)(rowptr + koff + 2 * hi8 + 8);
    h16v a;
#pragma unroll
    for (int i = 0; i < 8; ++i) { a[i] = lo[i]; a[i + 8] = hv[i]; }
    return a;
}

// ---------------------------------------------------------------------------
// Pack img/txt -> xp (f16), per reference: row s<il -> img[s]; else txt[clamp]
// ---------------------------------------------------------------------------
__global__ __launch_bounds__(256) void pack_x(const float* __restrict__ img,
                                              const float* __restrict__ txt,
                                              const int* __restrict__ il_,
                                              _Float16* __restrict__ xp) {
    int row = blockIdx.x;            // 0..MROWS-1
    int b = row / S_, s = row % S_;
    int il = il_[b];
    const float* src;
    if (s < il) {
        src = img + ((size_t)b * LIMG_ + s) * DIM_;
    } else {
        int t = s - il; if (t > LTXT_ - 1) t = LTXT_ - 1;
        src = txt + ((size_t)b * LTXT_ + t) * DIM_;
    }
    _Float16* dst = xp + (size_t)row * DIM_;
    for (int i = threadIdx.x; i < DIM_; i += blockDim.x)
        dst[i] = (_Float16)src[i];
}

__global__ __launch_bounds__(256) void cvt_f32_f16(const float* __restrict__ in,
                                                   _Float16* __restrict__ out, int n) {
    int i = blockIdx.x * blockDim.x + threadIdx.x;
    if (i < n) out[i] = (_Float16)in[i];
}

// ---------------------------------------------------------------------------
// QKV GEMM: C[4608,3072] = xp @ qkv_w^T ; scatter into q,k [bh][S][HD] and
// v transposed [bh][HD][S]. Wave computes 16x64; 4 waves per block.
// ---------------------------------------------------------------------------
__global__ __launch_bounds__(128) void gemm_qkv(const _Float16* __restrict__ xp,
                                                const _Float16* __restrict__ w,
                                                _Float16* __restrict__ q,
                                                _Float16* __restrict__ k,
                                                _Float16* __restrict__ vt) {
    const int lane = threadIdx.x & 31;
    const int wave = threadIdx.x >> 5;
    const int hi   = lane >> 4;      // 0/1
    const int hi8  = hi * 8;
    const int lN   = lane & 15;
    const int m0   = blockIdx.x * 64 + wave * 16;
    const int n0   = blockIdx.y * 64;

    const _Float16* arow = xp + (size_t)(m0 + lN) * DIM_;
    f8v acc[4] = {};
    for (int kk = 0; kk < DIM_; kk += 32) {
        h16v a = make_afrag(arow, kk, hi8);
#pragma unroll
        for (int t = 0; t < 4; ++t) {
            int n = n0 + t * 16 + lN;
            // B 32x16: lane holds col n; k = kk + 16*hi + e  (contiguous 16 halves)
            h16v bb = *(const h16v*)(w + (size_t)n * DIM_ + kk + 16 * hi);
            acc[t] = wmma_f16(a, bb, acc[t]);
        }
    }
#pragma unroll
    for (int t = 0; t < 4; ++t) {
        int n = n0 + t * 16 + lN;
        int which = n >> 10, d = n & 1023, head = d >> 6, hd = d & 63;
#pragma unroll
        for (int r = 0; r < 8; ++r) {
            int row = m0 + r + 8 * hi;
            int b = row / S_, s = row % S_;
            size_t bh = (size_t)b * NH_ + head;
            _Float16 hval = (_Float16)acc[t][r];
            if (which == 0)      q[(bh * S_ + s) * HD_ + hd] = hval;
            else if (which == 1) k[(bh * S_ + s) * HD_ + hd] = hval;
            else                 vt[(bh * HD_ + hd) * S_ + s] = hval;
        }
    }
}

// ---------------------------------------------------------------------------
// 2D RoPE on q,k image rows (s < il). One block per (b, s); thread = (head, pair).
// ---------------------------------------------------------------------------
__global__ __launch_bounds__(512) void rope_qk(_Float16* __restrict__ q,
                                               _Float16* __restrict__ k,
                                               const int* __restrict__ il_,
                                               const int* __restrict__ pos) {
    int row = blockIdx.x;                 // b*LIMG + s
    int b = row / LIMG_, s = row % LIMG_;
    if (s >= il_[b]) return;
    int t = threadIdx.x;
    int head = t >> 5;                    // 16 heads
    int pr = t & 31;
    int half = pr >> 4;                   // axis 0/1
    int i = pr & 15;                      // pair index within 32-wide half
    float p  = (float)pos[(size_t)row * 2 + half];
    float inv = __expf(-((float)(2 * i) / 32.0f) * 9.210340371976184f); // 10000^-(2i/32)
    float th = p * inv, c = __cosf(th), sn = __sinf(th);
    size_t base = ((size_t)(b * NH_ + head) * S_ + s) * HD_ + half * 32 + 2 * i;
    float q0 = (float)q[base], q1 = (float)q[base + 1];
    q[base]     = (_Float16)(q0 * c - q1 * sn);
    q[base + 1] = (_Float16)(q0 * sn + q1 * c);
    float k0 = (float)k[base], k1 = (float)k[base + 1];
    k[base]     = (_Float16)(k0 * c - k1 * sn);
    k[base + 1] = (_Float16)(k0 * sn + k1 * c);
}

// ---------------------------------------------------------------------------
// Flash attention: one wave per (bh, 16-query tile); keys in chunks of 32.
// ---------------------------------------------------------------------------
__global__ __launch_bounds__(128) void attn(const _Float16* __restrict__ q,
                                            const _Float16* __restrict__ k,
                                            const _Float16* __restrict__ vt,
                                            const int* __restrict__ il_,
                                            const int* __restrict__ tl_,
                                            _Float16* __restrict__ o) {
    __shared__ __align__(32) _Float16 plds[4][16 * 32];
    const int lane = threadIdx.x & 31;
    const int wave = threadIdx.x >> 5;
    const int hi = lane >> 4, hi8 = hi * 8, lN = lane & 15;
    int task = blockIdx.x * 4 + wave;
    int bh = task / QTILES;
    int qt = task % QTILES;
    int b  = bh / NH_;
    int h  = bh % NH_;
    int vlen = il_[b] + tl_[b];
    int q0 = qt * 16;

    const _Float16* qrow = q + ((size_t)bh * S_ + q0 + lN) * HD_;
    h16v aq[2];
#pragma unroll
    for (int c = 0; c < 2; ++c) aq[c] = make_afrag(qrow, 32 * c, hi8);

    f8v Oa[4] = {};
    float m[8], l[8];
#pragma unroll
    for (int r = 0; r < 8; ++r) { m[r] = -3.0e38f; l[r] = 0.0f; }
    _Float16* pl = plds[wave];

    for (int kj = 0; kj < S_; kj += 32) {
        f8v sacc[2] = {};
#pragma unroll
        for (int u = 0; u < 2; ++u) {
            int key = kj + u * 16 + lN;
            const _Float16* krow = k + ((size_t)bh * S_ + key) * HD_;
#pragma unroll
            for (int c = 0; c < 2; ++c) {
                // B 64x16 (K^T): lane holds key col; kk = 32c + 16*hi + e
                h16v bk = *(const h16v*)(krow + 32 * c + 16 * hi);
                sacc[u] = wmma_f16(aq[c], bk, sacc[u]);
            }
        }
        float sc[2][8];
#pragma unroll
        for (int u = 0; u < 2; ++u) {
            bool ok = (kj + u * 16 + lN) < vlen;
#pragma unroll
            for (int r = 0; r < 8; ++r)
                sc[u][r] = ok ? sacc[u][r] * SCALE_ : -1.0e30f;
        }
        float mn[8], al[8];
#pragma unroll
        for (int r = 0; r < 8; ++r) {
            float v = fmaxf(sc[0][r], sc[1][r]);
            v = fmaxf(v, __shfl_xor(v, 1, 32));
            v = fmaxf(v, __shfl_xor(v, 2, 32));
            v = fmaxf(v, __shfl_xor(v, 4, 32));
            v = fmaxf(v, __shfl_xor(v, 8, 32));
            mn[r] = fmaxf(m[r], v);
            al[r] = __expf(m[r] - mn[r]);
            m[r]  = mn[r];
        }
#pragma unroll
        for (int r = 0; r < 8; ++r) {
            float p0 = __expf(sc[0][r] - mn[r]);
            float p1 = __expf(sc[1][r] - mn[r]);
            int prow = (r + 8 * hi) * 32;
            pl[prow + lN]      = (_Float16)p0;      // D-layout -> LDS row-major
            pl[prow + 16 + lN] = (_Float16)p1;
            float rs = p0 + p1;
            rs += __shfl_xor(rs, 1, 32);
            rs += __shfl_xor(rs, 2, 32);
            rs += __shfl_xor(rs, 4, 32);
            rs += __shfl_xor(rs, 8, 32);
            l[r] = l[r] * al[r] + rs;
        }
#pragma unroll
        for (int t = 0; t < 4; ++t)
#pragma unroll
            for (int r = 0; r < 8; ++r) Oa[t][r] *= al[r];

        asm volatile("s_wait_dscnt 0" ::: "memory");   // LDS transpose fence
        h16v ap = make_afrag(pl + lN * 32, 0, hi8);    // P as A 16x32

#pragma unroll
        for (int t = 0; t < 4; ++t) {
            // V as B 32x16 from transposed store: contiguous along keys
            const _Float16* vrow = vt + ((size_t)bh * HD_ + t * 16 + lN) * S_ + kj + 16 * hi;
            h16v bv = *(const h16v*)vrow;
            Oa[t] = wmma_f16(ap, bv, Oa[t]);
        }
        asm volatile("" ::: "memory");
    }
#pragma unroll
    for (int r = 0; r < 8; ++r) l[r] = 1.0f / l[r];
#pragma unroll
    for (int t = 0; t < 4; ++t)
#pragma unroll
        for (int r = 0; r < 8; ++r) {
            size_t row = (size_t)b * S_ + q0 + r + 8 * hi;
            o[row * DIM_ + h * HD_ + t * 16 + lN] = (_Float16)(Oa[t][r] * l[r]);
        }
}

// ---------------------------------------------------------------------------
// Output projection + bias + validity mask + img/txt scatter.
// ---------------------------------------------------------------------------
__global__ __launch_bounds__(128) void gemm_out(const _Float16* __restrict__ o16,
                                                const _Float16* __restrict__ w,
                                                const float* __restrict__ bias,
                                                const int* __restrict__ il_,
                                                const int* __restrict__ tl_,
                                                float* __restrict__ out) {
    const int lane = threadIdx.x & 31;
    const int wave = threadIdx.x >> 5;
    const int hi = lane >> 4, hi8 = hi * 8, lN = lane & 15;
    const int m0 = blockIdx.x * 64 + wave * 16;
    const int n0 = blockIdx.y * 64;

    const _Float16* arow = o16 + (size_t)(m0 + lN) * DIM_;
    f8v acc[4] = {};
    for (int kk = 0; kk < DIM_; kk += 32) {
        h16v a = make_afrag(arow, kk, hi8);
#pragma unroll
        for (int t = 0; t < 4; ++t) {
            int n = n0 + t * 16 + lN;
            h16v bb = *(const h16v*)(w + (size_t)n * DIM_ + kk + 16 * hi);
            acc[t] = wmma_f16(a, bb, acc[t]);
        }
    }
#pragma unroll
    for (int t = 0; t < 4; ++t) {
        int n = n0 + t * 16 + lN;
        float bn = bias[n];
#pragma unroll
        for (int r = 0; r < 8; ++r) {
            int row = m0 + r + 8 * hi;
            int b = row / S_, s = row % S_;
            int il = il_[b], tl = tl_[b];
            float v = acc[t][r] + bn;
            if (s < il) {
                out[((size_t)b * LIMG_ + s) * DIM_ + n] = v;
            } else {
                int ts = s - il;
                if (ts < tl)
                    out[(size_t)B_ * LIMG_ * DIM_ + ((size_t)b * LTXT_ + ts) * DIM_ + n] = v;
            }
        }
    }
}

extern "C" void kernel_launch(void* const* d_in, const int* in_sizes, int n_in,
                              void* d_out, int out_size, void* d_ws, size_t ws_size,
                              hipStream_t stream) {
    const float* img   = (const float*)d_in[0];
    const float* txt   = (const float*)d_in[1];
    const float* qkv_w = (const float*)d_in[2];
    const float* out_w = (const float*)d_in[3];
    const float* out_b = (const float*)d_in[4];
    const int*   il    = (const int*)d_in[5];
    const int*   tl    = (const int*)d_in[6];
    const int*   pos   = (const int*)d_in[7];
    float* out = (float*)d_out;

    char* ws = (char*)d_ws;
    size_t off = 0;
    auto take = [&](size_t bytes) -> void* {
        void* p = ws + off;
        off += (bytes + 255) & ~(size_t)255;
        return p;
    };
    _Float16* xp  = (_Float16*)take((size_t)MROWS * DIM_ * 2);       // packed input f16
    _Float16* wq  = (_Float16*)take((size_t)3 * DIM_ * DIM_ * 2);    // qkv_w f16
    _Float16* wo  = (_Float16*)take((size_t)DIM_ * DIM_ * 2);        // out_w f16
    _Float16* qb  = (_Float16*)take((size_t)B_ * NH_ * S_ * HD_ * 2);
    _Float16* kb  = (_Float16*)take((size_t)B_ * NH_ * S_ * HD_ * 2);
    _Float16* vtb = (_Float16*)take((size_t)B_ * NH_ * S_ * HD_ * 2); // V transposed [bh][HD][S]
    _Float16* o16 = (_Float16*)take((size_t)MROWS * DIM_ * 2);       // attention output f16
    (void)ws_size; (void)in_sizes; (void)n_in;

    hipMemsetAsync(d_out, 0, (size_t)out_size * sizeof(float), stream);

    pack_x<<<MROWS, 256, 0, stream>>>(img, txt, il, xp);
    cvt_f32_f16<<<(3 * DIM_ * DIM_ + 255) / 256, 256, 0, stream>>>(qkv_w, wq, 3 * DIM_ * DIM_);
    cvt_f32_f16<<<(DIM_ * DIM_ + 255) / 256, 256, 0, stream>>>(out_w, wo, DIM_ * DIM_);

    gemm_qkv<<<dim3(MROWS / 64, (3 * DIM_) / 64), 128, 0, stream>>>(xp, wq, qb, kb, vtb);
    rope_qk<<<B_ * LIMG_, 512, 0, stream>>>(qb, kb, il, pos);
    attn<<<(B_ * NH_ * QTILES) / 4, 128, 0, stream>>>(qb, kb, vtb, il, tl, o16);
    gemm_out<<<dim3(MROWS / 64, DIM_ / 64), 128, 0, stream>>>(o16, wo, out_b, il, tl, out);
}
